// Encoder_31018253812370
// MI455X (gfx1250) — matrix operand order repaired and checked by reading, not compiled, not run on hardware
//
#include <hip/hip_runtime.h>

// ---------------- problem constants ----------------
#define T_SEQ 4096
#define D_IN  1024
#define H_DIR 512
#define G4H   2048            // 4*H_DIR gate rows
#define NB_PER_DIR 16         // recurrence blocks per direction
#define UNITS_PER_BLK 32      // H_DIR / NB_PER_DIR
#define ROWS_PER_BLK 128      // 4 gates * UNITS_PER_BLK

typedef __attribute__((ext_vector_type(16))) _Float16 v16h;
typedef __attribute__((ext_vector_type(8)))  _Float16 v8h;
typedef __attribute__((ext_vector_type(8)))  float    v8f;

// ============================================================
// f32 -> f16 pack (one-time; GEMM then streams f16 directly)
// ============================================================
__global__ __launch_bounds__(256)
void pack_f16(const float* __restrict__ src, _Float16* __restrict__ dst, int n)
{
    const int i = (blockIdx.x * 256 + threadIdx.x) * 8;
    if (i + 8 <= n) {
#pragma unroll
        for (int q = 0; q < 8; ++q) dst[i + q] = (_Float16)src[i + q];
    }
}

// ============================================================
// GEMM: Gx[t, r] = sum_k X[t,k] * W[r,k] + bih[r] + bhh[r]
// Xh [T x K] f16 row-major (optionally row-reversed), Wh [2048 x K] f16.
// Block: 256 threads (8 waves). Wave tile 16x128 (8 WMMAs / k-chunk,
// one A-fragment amortized over 8 B-fragments).
// Block tile 128x128. grid = (2048/128, 4096/128, 2 dirs).
// ============================================================
__global__ __launch_bounds__(256)
void lstm_gate_gemm(const _Float16* __restrict__ X0, const _Float16* __restrict__ X1,
                    const _Float16* __restrict__ W0, const _Float16* __restrict__ W1,
                    const float* __restrict__ bi0, const float* __restrict__ bh0,
                    const float* __restrict__ bi1, const float* __restrict__ bh1,
                    float* __restrict__ Gx0, float* __restrict__ Gx1,
                    int K, int revMask)
{
    const int dir = blockIdx.z;
    const _Float16* __restrict__ X = dir ? X1 : X0;
    const _Float16* __restrict__ W = dir ? W1 : W0;
    const float* __restrict__ bi = dir ? bi1 : bi0;
    const float* __restrict__ bh = dir ? bh1 : bh0;
    float* __restrict__ Gx       = dir ? Gx1 : Gx0;
    const bool rev = (revMask >> dir) & 1;

    const int lane = threadIdx.x & 31;
    const int wave = threadIdx.x >> 5;
    const int m0 = blockIdx.y * 128 + wave * 16;   // 8 waves -> 128 rows
    const int n0 = blockIdx.x * 128;               // each wave covers 128 cols

    // A-fragment addressing (16x32 f16, 16 contiguous K per lane half, same order as B)
    const int mrow = m0 + (lane & 15);
    const int arow = rev ? (T_SEQ - 1 - mrow) : mrow;
    const _Float16* __restrict__ xr = X + (size_t)arow * K;
    const int ka = (lane >> 4) * 16;  // lanes 0-15: K 0..15 ; 16-31: 16..31

    // B-fragment addressing (32x16: column n = lane&15, 16 contiguous K per lane half)
    const int bn = lane & 15;
    const int kb = (lane >> 4) * 16;

    v8f acc[8] = {};

    for (int kc = 0; kc < K; kc += 32) {
        const v8h alo = *(const v8h*)(xr + kc + ka);
        const v8h ahi = *(const v8h*)(xr + kc + ka + 8);
        const v16h a = __builtin_shufflevector(alo, ahi,
                                               0, 1, 2, 3, 4, 5, 6, 7,
                                               8, 9, 10, 11, 12, 13, 14, 15);
#pragma unroll
        for (int nt = 0; nt < 8; ++nt) {
            const _Float16* __restrict__ wp =
                W + (size_t)(n0 + nt * 16 + bn) * K + kc + kb;
            const v16h b = *(const v16h*)wp;
            acc[nt] = __builtin_amdgcn_wmma_f32_16x16x32_f16(
                false, a, false, b, (short)0, acc[nt], false, false);
        }
    }

    // C/D layout: VGPR v -> M = m0 + v + 8*(lane>=16), N = n0 + (lane&15)
    const int mbase = m0 + (lane >> 4) * 8;
#pragma unroll
    for (int nt = 0; nt < 8; ++nt) {
        const int ncol = n0 + nt * 16 + (lane & 15);
        const float bias = bi[ncol] + bh[ncol];
#pragma unroll
        for (int v = 0; v < 8; ++v) {
            Gx[(size_t)(mbase + v) * G4H + ncol] = acc[nt][v] + bias;
        }
    }
}

// ============================================================
// Recurrence: 16 blocks/direction (32 total), each block pins a
// 256 KB slice of Whh (32 units x 4 gates x 512 f32) in LDS.
// Per step: LDS matvec -> gates -> c/h update -> h exchange through
// a device-scope double-buffered h vector + atomic grid barrier.
// Layer 0 writes its output sequence as f16 (layer-1 GEMM A-matrix).
// ============================================================
__global__ __launch_bounds__(1024)
void lstm_recurrence(const float* __restrict__ Gx0, const float* __restrict__ Gx1,
                     const float* __restrict__ Whh0, const float* __restrict__ Whh1,
                     _Float16* __restrict__ y16_0, _Float16* __restrict__ y16_1,
                     float* __restrict__ hbuf0, float* __restrict__ hbuf1,
                     unsigned* __restrict__ ctr0, unsigned* __restrict__ ctr1,
                     float* __restrict__ out_final, int layer)
{
    const int dir   = blockIdx.x / NB_PER_DIR;
    const int slice = blockIdx.x % NB_PER_DIR;
    const float* __restrict__ Gx  = dir ? Gx1  : Gx0;
    const float* __restrict__ Whh = dir ? Whh1 : Whh0;
    _Float16* __restrict__ y16    = dir ? y16_1 : y16_0;
    float* __restrict__ hbuf      = dir ? hbuf1 : hbuf0;
    unsigned* __restrict__ ctr    = dir ? ctr1  : ctr0;
    const int u0 = slice * UNITS_PER_BLK;

    __shared__ float whh_s[4][UNITS_PER_BLK][H_DIR]; // 256 KB resident weights
    __shared__ float h_s[H_DIR];
    __shared__ float part_s[ROWS_PER_BLK][8];
    __shared__ float gpre_s[ROWS_PER_BLK];

    const int tid = threadIdx.x;

    // stage Whh slice into LDS once
    for (int idx = tid; idx < ROWS_PER_BLK * H_DIR; idx += 1024) {
        const int k   = idx & (H_DIR - 1);
        const int row = idx >> 9;
        const int g   = row >> 5;
        const int j   = row & 31;
        whh_s[g][j][k] = Whh[(size_t)(g * H_DIR + u0 + j) * H_DIR + k];
    }
    for (int i = tid; i < H_DIR; i += 1024) h_s[i] = 0.f;

    float c = 0.f;
    __syncthreads();

    const int row = tid >> 3;            // 0..127 gate row within slice
    const int seg = tid & 7;             // k-segment
    const int g = row >> 5, j = row & 31;
    const int kbase = seg * 64;

    for (int t = 0; t < T_SEQ; ++t) {
        // partial dot products from LDS-resident weights
        float p = 0.f;
        const float* __restrict__ wr = &whh_s[g][j][kbase];
        const float* __restrict__ hr = &h_s[kbase];
#pragma unroll 16
        for (int k = 0; k < 64; ++k) p += wr[k] * hr[k];
        part_s[row][seg] = p;
        __syncthreads();

        if (tid < ROWS_PER_BLK) {
            const size_t gcol = (size_t)(tid >> 5) * H_DIR + u0 + (tid & 31);
            float s = Gx[(size_t)t * G4H + gcol];
#pragma unroll
            for (int q = 0; q < 8; ++q) s += part_s[tid][q];
            gpre_s[tid] = s;
            if (t + 1 < T_SEQ)   // pull next step's gate row toward the WGP
                __builtin_prefetch(&Gx[(size_t)(t + 1) * G4H + gcol], 0, 3);
        }
        __syncthreads();

        if (tid < UNITS_PER_BLK) {
            const float gi = gpre_s[tid];
            const float gf = gpre_s[32 + tid];
            const float gg = gpre_s[64 + tid];
            const float go = gpre_s[96 + tid];
            const float i_ = 1.f / (1.f + __expf(-gi));
            const float f_ = 1.f / (1.f + __expf(-gf));
            const float g_ = tanhf(gg);
            const float o_ = 1.f / (1.f + __expf(-go));
            c = f_ * c + i_ * g_;
            const float h = o_ * tanhf(c);

            __hip_atomic_store(&hbuf[(t & 1) * H_DIR + u0 + tid], h,
                               __ATOMIC_RELAXED, __HIP_MEMORY_SCOPE_AGENT);
            if (layer == 0) {
                y16[(size_t)t * H_DIR + u0 + tid] = (_Float16)h;  // layer-1 GEMM input
            } else {
                const int trow = dir ? (T_SEQ - 1 - t) : t;
                out_final[4096 + (size_t)trow * 1024 + dir * H_DIR + u0 + tid] = h;
            }
            if (t == T_SEQ - 1) {
                out_final[layer * 1024 + dir * H_DIR + u0 + tid] = c;          // cells
                out_final[2048 + layer * 1024 + dir * H_DIR + u0 + tid] = h;   // hiddens
            }
        }

        // device-scope barrier across the 16 blocks of this direction
        __threadfence();
        __syncthreads();
        if (tid == 0) {
            __hip_atomic_fetch_add(ctr, 1u, __ATOMIC_RELEASE, __HIP_MEMORY_SCOPE_AGENT);
            const unsigned target = (unsigned)(NB_PER_DIR * (t + 1));
            while (__hip_atomic_load(ctr, __ATOMIC_ACQUIRE, __HIP_MEMORY_SCOPE_AGENT) < target)
                __builtin_amdgcn_s_sleep(2);
        }
        __syncthreads();
        __threadfence();

        // reload full h vector (agent-scope loads bypass stale near caches)
        for (int i = tid; i < H_DIR; i += 1024)
            h_s[i] = __hip_atomic_load(&hbuf[(t & 1) * H_DIR + i],
                                       __ATOMIC_RELAXED, __HIP_MEMORY_SCOPE_AGENT);
        __syncthreads();
    }
}

// ============================================================
// host launch
// ============================================================
extern "C" void kernel_launch(void* const* d_in, const int* in_sizes, int n_in,
                              void* d_out, int out_size, void* d_ws, size_t ws_size,
                              hipStream_t stream) {
    (void)in_sizes; (void)n_in; (void)out_size; (void)ws_size;

    const float* emb   = (const float*)d_in[0];
    const float* fWih0 = (const float*)d_in[1];
    const float* fWhh0 = (const float*)d_in[2];
    const float* fbih0 = (const float*)d_in[3];
    const float* fbhh0 = (const float*)d_in[4];
    const float* fWih1 = (const float*)d_in[5];
    const float* fWhh1 = (const float*)d_in[6];
    const float* fbih1 = (const float*)d_in[7];
    const float* fbhh1 = (const float*)d_in[8];
    const float* bWih0 = (const float*)d_in[9];
    const float* bWhh0 = (const float*)d_in[10];
    const float* bbih0 = (const float*)d_in[11];
    const float* bbhh0 = (const float*)d_in[12];
    const float* bWih1 = (const float*)d_in[13];
    const float* bWhh1 = (const float*)d_in[14];
    const float* bbih1 = (const float*)d_in[15];
    const float* bbhh1 = (const float*)d_in[16];
    float* out = (float*)d_out;

    char* ws = (char*)d_ws;
    const size_t szGx  = (size_t)T_SEQ * G4H * sizeof(float);       // 32 MB
    const size_t szXh  = (size_t)T_SEQ * D_IN * sizeof(_Float16);   // 8 MB
    const size_t szYh  = (size_t)T_SEQ * H_DIR * sizeof(_Float16);  // 4 MB
    const size_t szW0h = (size_t)G4H * D_IN * sizeof(_Float16);     // 4 MB
    const size_t szW1h = (size_t)G4H * H_DIR * sizeof(_Float16);    // 2 MB

    size_t off = 0;
    float* GxF = (float*)(ws + off); off += szGx;
    float* GxB = (float*)(ws + off); off += szGx;
    _Float16* embH = (_Float16*)(ws + off); off += szXh;
    _Float16* y0Fh = (_Float16*)(ws + off); off += szYh;
    _Float16* y0Bh = (_Float16*)(ws + off); off += szYh;
    _Float16* WhF0 = (_Float16*)(ws + off); off += szW0h;
    _Float16* WhB0 = (_Float16*)(ws + off); off += szW0h;
    _Float16* WhF1 = (_Float16*)(ws + off); off += szW1h;
    _Float16* WhB1 = (_Float16*)(ws + off); off += szW1h;
    float* hbF = (float*)(ws + off); off += 2 * H_DIR * sizeof(float);
    float* hbB = (float*)(ws + off); off += 2 * H_DIR * sizeof(float);
    unsigned* ctrs = (unsigned*)(ws + off);

    const dim3 pblk(256);
    const int nEmb = T_SEQ * D_IN, nW0 = G4H * D_IN, nW1 = G4H * H_DIR;
    pack_f16<<<dim3(nEmb / (256 * 8)), pblk, 0, stream>>>(emb,   embH, nEmb);
    pack_f16<<<dim3(nW0  / (256 * 8)), pblk, 0, stream>>>(fWih0, WhF0, nW0);
    pack_f16<<<dim3(nW0  / (256 * 8)), pblk, 0, stream>>>(bWih0, WhB0, nW0);
    pack_f16<<<dim3(nW1  / (256 * 8)), pblk, 0, stream>>>(fWih1, WhF1, nW1);
    pack_f16<<<dim3(nW1  / (256 * 8)), pblk, 0, stream>>>(bWih1, WhB1, nW1);

    const dim3 gblk(256);
    const dim3 ggrid(G4H / 128, T_SEQ / 128, 2);

    // ---- layer 0 ----
    hipMemsetAsync(ctrs, 0, 2 * sizeof(unsigned), stream);
    lstm_gate_gemm<<<ggrid, gblk, 0, stream>>>(
        embH, embH, WhF0, WhB0, fbih0, fbhh0, bbih0, bbhh0,
        GxF, GxB, D_IN, /*revMask: bwd reads emb reversed*/ 0x2);
    lstm_recurrence<<<2 * NB_PER_DIR, 1024, 0, stream>>>(
        GxF, GxB, fWhh0, bWhh0, y0Fh, y0Bh, hbF, hbB, ctrs, ctrs + 1, out, 0);

    // ---- layer 1 ----
    hipMemsetAsync(ctrs, 0, 2 * sizeof(unsigned), stream);
    lstm_gate_gemm<<<ggrid, gblk, 0, stream>>>(
        y0Fh, y0Bh, WhF1, WhB1, fbih1, fbhh1, bbih1, bbhh1,
        GxF, GxB, H_DIR, 0x0);
    lstm_recurrence<<<2 * NB_PER_DIR, 1024, 0, stream>>>(
        GxF, GxB, fWhh1, bWhh1, y0Fh, y0Bh, hbF, hbB, ctrs, ctrs + 1, out, 1);
}
